// CellGraphTransformer_2241972928605
// MI455X (gfx1250) — matrix operand order repaired and checked by reading, not compiled, not run on hardware
//
#include <hip/hip_runtime.h>

// ---------------------------------------------------------------------------
// CellGraphTransformer for MI455X (gfx1250, wave32, WMMA 16x16x32 f16).
// All GEMMs run on v_wmma_f32_16x16x32_f16 with f32 accumulation.
// Each GEMM wave computes a 16x64 output strip: 4 independent WMMAs per
// k-step sharing one A fragment (XDL-friendly, 4x A reuse).
// Attention stages its 16KB bias tile via GLOBAL_LOAD_ASYNC_TO_LDS_B128,
// overlapped with the QK WMMA phase, synchronized with s_wait_asynccnt.
// Entire working set (~110 MB) is L2-resident on MI455X (192 MB L2).
// ---------------------------------------------------------------------------

typedef __attribute__((ext_vector_type(16))) _Float16 v16h;
typedef __attribute__((ext_vector_type(8)))  float    v8f;

#define BB   32          // batch
#define NNc  255         // cells per board
#define DD   256         // model dim
#define HH   8           // heads
#define HDD  32          // head dim
#define FFD  512         // ffn dim
#define LL   6           // layers
#define SS   256         // sequence length (NNc + 1)
#define MM   (BB * SS)   // 8192 total rows
#define TN   (BB * NNc)  // 8160 tokens
#define EE   (TN * 8)    // 65280 edges
#define LWH  524288      // f16 elements of transposed weights per layer

// --- Fragment loaders (wave32), contiguous k order used by both A and B -------------------------
// A (16x32 f16, row-major, row = lane&15):
//   lanes 0-15 : halves 0..15 = K 0..15
//   lanes16-31 : halves 0..15 = K 16..31
static __device__ __forceinline__ v16h load_afrag(const _Float16* __restrict__ rowp, int sel) {
  v16h f;
  const int kb = sel * 16;
#pragma unroll
  for (int i = 0; i < 16; ++i) f[i] = rowp[kb + i];
  return f;
}
// B (32x16 f16, per lane = one column n = lane&15 of B, i.e. one contiguous
// K-run of the transposed matrix): lanes 0-15 K 0..15, lanes 16-31 K 16..31.
static __device__ __forceinline__ v16h load_bfrag(const _Float16* __restrict__ colp, int sel) {
  v16h f;
  const int kb = sel * 16;
#pragma unroll
  for (int i = 0; i < 16; ++i) f[i] = colp[kb + i];
  return f;
}

// --- weight transpose + f32->f16 convert: dst[n*K + k] = src[k*N + n] -------
__global__ __launch_bounds__(256)
void transpose_cvt(const float* __restrict__ src, _Float16* __restrict__ dst, int K, int N) {
  int idx = blockIdx.x * 256 + threadIdx.x;
  if (idx >= K * N) return;
  int k = idx / N, n = idx - k * N;
  dst[(size_t)n * K + k] = (_Float16)src[idx];
}

// --- x0 = concat(game_token, patch_emb[cell_states]) ------------------------
__global__ __launch_bounds__(256)
void embed_kernel(const int* __restrict__ cs, const float* __restrict__ pe,
                  const float* __restrict__ gt, float* __restrict__ x) {
  int row = blockIdx.x, d = threadIdx.x;
  int b = row >> 8, s = row & 255;
  float v = (s == 0) ? gt[d] : pe[(size_t)cs[b * NNc + (s - 1)] * DD + d];
  x[(size_t)row * DD + d] = v;
}

__global__ __launch_bounds__(256)
void zero_f32(float* __restrict__ p, int n) {
  int i = blockIdx.x * 256 + threadIdx.x;
  if (i < n) p[i] = 0.0f;
}

// --- bias[bi, :, 1+src%N, 1+dst%N] = edge_emb[edge_type] --------------------
__global__ __launch_bounds__(256)
void edge_scatter(const int* __restrict__ ei, const int* __restrict__ et,
                  const int* __restrict__ bv, const float* __restrict__ eemb,
                  float* __restrict__ bias) {
  int e = blockIdx.x * 256 + threadIdx.x;
  if (e >= EE) return;
  int src = ei[e], dst = ei[EE + e];
  int bi = bv[src];
  int r = src % NNc + 1, c = dst % NNc + 1;
  const float* vals = eemb + (size_t)et[e] * HH;
#pragma unroll
  for (int h = 0; h < HH; ++h)
    bias[(((size_t)(bi * HH + h)) * SS + r) * SS + c] = vals[h];
}

// --- LayerNorm (one block = one row of D=256) -> f16 ------------------------
__global__ __launch_bounds__(256)
void ln_to_f16(const float* __restrict__ x, const float* __restrict__ g,
               const float* __restrict__ bta, _Float16* __restrict__ outH) {
  __shared__ float red[256];
  const int row = blockIdx.x, t = threadIdx.x;
  const float v = x[(size_t)row * DD + t];
  red[t] = v;
  __syncthreads();
  for (int s = 128; s > 0; s >>= 1) {
    if (t < s) red[t] += red[t + s];
    __syncthreads();
  }
  const float mean = red[0] * (1.0f / DD);
  __syncthreads();
  const float d = v - mean;
  red[t] = d * d;
  __syncthreads();
  for (int s = 128; s > 0; s >>= 1) {
    if (t < s) red[t] += red[t + s];
    __syncthreads();
  }
  const float var = red[0] * (1.0f / DD);
  const float o = d * rsqrtf(var + 1e-5f) * g[t] + bta[t];
  outH[(size_t)row * DD + t] = (_Float16)o;
}

// --- WMMA GEMM: each wave computes a 16x64 strip (4 accumulators) -----------
// Block = 4 waves = 64 rows x 64 cols per block.
// MODE 0: QKV epilogue  -> scatter Q/K as (b,h,s,hd) f16, V as (b,h,hd,s) f16
// MODE 1: residual      -> xres[row*ldout+col] += acc + bias[col]
// MODE 2: bias+ReLU f16 -> outH[row*ldout+col]  = relu(acc + bias[col])
template <int MODE>
__global__ __launch_bounds__(128)
void gemm16(const _Float16* __restrict__ A, const _Float16* __restrict__ WT,
            const float* __restrict__ bvec, int K, int ldout,
            float* __restrict__ xres, _Float16* __restrict__ outH,
            _Float16* __restrict__ Qh, _Float16* __restrict__ Kh,
            _Float16* __restrict__ Vt) {
  const int wave = threadIdx.x >> 5;
  const int lane = threadIdx.x & 31;
  const int n = lane & 15, sel = lane >> 4;
  const int row0 = (blockIdx.y << 6) + (wave << 4);
  const int col0 = blockIdx.x << 6;
  const _Float16* ar = A + (size_t)(row0 + n) * K;
  const _Float16* br = WT + (size_t)(col0 + n) * K;

  // hint the 4 weight rows this lane will stream (global_prefetch_b8)
  __builtin_prefetch(br, 0, 1);
  __builtin_prefetch(br + (size_t)16 * K, 0, 1);
  __builtin_prefetch(br + (size_t)32 * K, 0, 1);
  __builtin_prefetch(br + (size_t)48 * K, 0, 1);

  v8f acc[4] = {};
  for (int k0 = 0; k0 < K; k0 += 32) {
    v16h af = load_afrag(ar + k0, sel);
#pragma unroll
    for (int t = 0; t < 4; ++t) {
      v16h bf = load_bfrag(br + (size_t)(t * 16) * K + k0, sel);
      acc[t] = __builtin_amdgcn_wmma_f32_16x16x32_f16(false, af, false, bf,
                                                      (short)0, acc[t], false, false);
    }
  }

#pragma unroll
  for (int t = 0; t < 4; ++t) {
    const int col = col0 + t * 16 + n;
    const float bv = bvec[col];
#pragma unroll
    for (int j = 0; j < 8; ++j) {
      const int row = row0 + j + (sel << 3);
      float v = acc[t][j] + bv;
      if (MODE == 0) {
        const int b = row >> 8, s = row & 255;
        const int which = col >> 8, within = col & 255;
        const int h = within >> 5, hd = within & 31;
        const size_t bh = (size_t)(b * HH + h);
        if (which == 0)      Qh[(bh * SS + s) * HDD + hd] = (_Float16)v;
        else if (which == 1) Kh[(bh * SS + s) * HDD + hd] = (_Float16)v;
        else                 Vt[(bh * HDD + hd) * SS + s] = (_Float16)v;
      } else if (MODE == 1) {
        xres[(size_t)row * ldout + col] += v;
      } else {
        outH[(size_t)row * ldout + col] = (_Float16)fmaxf(v, 0.0f);
      }
    }
  }
}

// --- fused attention: one wave handles 16 queries of one (b,h) --------------
// Phase 0: kick off async copy of the contiguous 16KB bias tile
//          (rows q0..q0+15 of bias[b,h]) into LDS via the ASYNC engine.
// Phase 1: Q.K^T -- one 16x16x32 WMMA per key tile (16 independent).
// Phase 2: s_wait_asynccnt 0, softmax with bias/scale out of LDS.
// Phase 3: P.V with 2 interleaved accumulators sharing each P fragment.
__global__ __launch_bounds__(32)
void attn_kernel(const _Float16* __restrict__ Qh, const _Float16* __restrict__ Kh,
                 const _Float16* __restrict__ Vt, const float* __restrict__ bias,
                 _Float16* __restrict__ Oh) {
  __shared__ float    sc[16 * SS];
  __shared__ _Float16 sch[16 * SS];
  __shared__ float    sbias[16 * SS];
  const int lane = threadIdx.x & 31;
  const int n = lane & 15, sel = lane >> 4;
  const int q0 = blockIdx.x << 4;
  const int h = blockIdx.y, b = blockIdx.z;
  const size_t bh = (size_t)(b * HH + h);

  // async global->LDS copy of bias rows q0..q0+15 (contiguous 16KB).
  // GV mode: INST_OFFSET is added to both the LDS address in VDST and the
  // global address in VADDR, so one per-lane base covers the whole tile.
  {
    const uint8_t* gsrc = (const uint8_t*)(bias + (bh * SS + q0) * SS) + lane * 16;
    uint32_t ldst = (uint32_t)(uintptr_t)(&sbias[0]) + lane * 16;
#pragma unroll
    for (int i = 0; i < 32; ++i)  // 32 lanes x 16B x 32 ops = 16KB
      asm volatile("global_load_async_to_lds_b128 %0, %1, off offset:%2"
                   :: "v"(ldst), "v"(gsrc), "i"(i * 512) : "memory");
  }

  // scores = Q (16x32) x K^T (32x16 per tile); Q fragment is loop-invariant
  const v16h qf = load_afrag(Qh + (bh * SS + q0 + n) * HDD, sel);
  for (int nt = 0; nt < 16; ++nt) {
    v16h kf = load_bfrag(Kh + (bh * SS + nt * 16 + n) * HDD, sel);
    v8f acc = {};
    acc = __builtin_amdgcn_wmma_f32_16x16x32_f16(false, qf, false, kf,
                                                 (short)0, acc, false, false);
#pragma unroll
    for (int j = 0; j < 8; ++j)
      sc[(j + (sel << 3)) * SS + nt * 16 + n] = acc[j];
  }
  __syncthreads();
  asm volatile("s_wait_asynccnt 0" ::: "memory");  // bias tile landed in LDS

  // softmax over 256 keys, one lane per query row (bias read from LDS)
  if (lane < 16) {
    const float* brow = sbias + lane * SS;
    float* srow = sc + lane * SS;
    float mx = -1e30f;
    for (int c = 0; c < SS; ++c) {
      float tv = srow[c] * 0.17677669529663687f + brow[c];  // 1/sqrt(32)
      srow[c] = tv;
      mx = fmaxf(mx, tv);
    }
    float sum = 0.0f;
    for (int c = 0; c < SS; ++c) {
      float e = __expf(srow[c] - mx);
      srow[c] = e;
      sum += e;
    }
    const float inv = 1.0f / sum;
    _Float16* hrow = sch + lane * SS;
    for (int c = 0; c < SS; ++c) hrow[c] = (_Float16)(srow[c] * inv);
  }
  __syncthreads();

  // O = P (16x256) x V (256x32); V stored transposed (hd, s)
  const _Float16* pr  = sch + n * SS;
  const _Float16* vr0 = Vt + (bh * HDD + n) * SS;       // hd column tile 0
  const _Float16* vr1 = Vt + (bh * HDD + 16 + n) * SS;  // hd column tile 1
  v8f acc0 = {}, acc1 = {};
  for (int kt = 0; kt < 8; ++kt) {
    v16h pf = load_afrag(pr + kt * 32, sel);
    v16h vf0 = load_bfrag(vr0 + kt * 32, sel);
    v16h vf1 = load_bfrag(vr1 + kt * 32, sel);
    acc0 = __builtin_amdgcn_wmma_f32_16x16x32_f16(false, pf, false, vf0,
                                                  (short)0, acc0, false, false);
    acc1 = __builtin_amdgcn_wmma_f32_16x16x32_f16(false, pf, false, vf1,
                                                  (short)0, acc1, false, false);
  }
#pragma unroll
  for (int j = 0; j < 8; ++j) {
    const size_t orow = (size_t)(b * SS + q0 + j + (sel << 3)) * DD + h * HDD;
    Oh[orow + n]      = (_Float16)acc0[j];
    Oh[orow + 16 + n] = (_Float16)acc1[j];
  }
}

// --- policy/value head (tiny; one block) ------------------------------------
__global__ __launch_bounds__(256)
void head_kernel(const float* __restrict__ x, const float* __restrict__ fc1_w,
                 const float* __restrict__ fc1_b, const float* __restrict__ pol_w,
                 const float* __restrict__ pol_b, const float* __restrict__ val_w,
                 const float* __restrict__ val_b, float* __restrict__ out) {
  __shared__ float hid[BB * 64];
  const int t = threadIdx.x;
  for (int idx = t; idx < BB * 64; idx += 256) {
    const int b = idx >> 6, j = idx & 63;
    const float* g = x + (size_t)(b * SS) * DD;  // row s=0 of board b
    float s = fc1_b[j];
    for (int k = 0; k < DD; ++k) s += g[k] * fc1_w[k * 64 + j];
    hid[idx] = fmaxf(s, 0.0f);
  }
  __syncthreads();
  if (t < BB * 7) {
    const int b = t / 7, o = t % 7;
    float s = pol_b[o];
    for (int j = 0; j < 64; ++j) s += hid[b * 64 + j] * pol_w[j * 7 + o];
    out[b * 7 + o] = s;
  } else if (t < BB * 7 + BB) {
    const int b = t - BB * 7;
    float s = val_b[0];
    for (int j = 0; j < 64; ++j) s += hid[b * 64 + j] * val_w[j];
    out[BB * 7 + b] = tanhf(s);
  }
}

// ---------------------------------------------------------------------------
extern "C" void kernel_launch(void* const* d_in, const int* in_sizes, int n_in,
                              void* d_out, int out_size, void* d_ws, size_t ws_size,
                              hipStream_t stream) {
  (void)in_sizes; (void)n_in; (void)out_size; (void)ws_size;
  const int*   cell_states = (const int*)d_in[0];
  const int*   edge_index  = (const int*)d_in[1];
  const int*   edge_type   = (const int*)d_in[2];
  const int*   batch_vec   = (const int*)d_in[3];
  const float* patch_emb   = (const float*)d_in[5];
  const float* game_token  = (const float*)d_in[6];
  const float* edge_emb    = (const float*)d_in[7];
  const float* in_w   = (const float*)d_in[8];
  const float* in_b   = (const float*)d_in[9];
  const float* out_w  = (const float*)d_in[10];
  const float* out_b  = (const float*)d_in[11];
  const float* ln1_g  = (const float*)d_in[12];
  const float* ln1_b  = (const float*)d_in[13];
  const float* ln2_g  = (const float*)d_in[14];
  const float* ln2_b  = (const float*)d_in[15];
  const float* ffn_w1 = (const float*)d_in[16];
  const float* ffn_b1 = (const float*)d_in[17];
  const float* ffn_w2 = (const float*)d_in[18];
  const float* ffn_b2 = (const float*)d_in[19];
  const float* fc1_w  = (const float*)d_in[20];
  const float* fc1_b  = (const float*)d_in[21];
  const float* pol_w  = (const float*)d_in[22];
  const float* pol_b  = (const float*)d_in[23];
  const float* val_w  = (const float*)d_in[24];
  const float* val_b  = (const float*)d_in[25];
  float* out = (float*)d_out;

  // workspace carve-up (~106 MB total)
  uint8_t* wp = (uint8_t*)d_ws;
  size_t off = 0;
  auto carve = [&](size_t bytes) -> void* {
    void* p = wp + off;
    off += (bytes + 255) & ~(size_t)255;
    return p;
  };
  float*    x    = (float*)carve((size_t)MM * DD * 4);            // activations f32
  float*    bias = (float*)carve((size_t)BB * HH * SS * SS * 4);  // attn bias
  _Float16* a_h  = (_Float16*)carve((size_t)MM * DD * 2);         // LN output f16
  _Float16* o_h  = (_Float16*)carve((size_t)MM * DD * 2);         // attn out f16
  _Float16* fh   = (_Float16*)carve((size_t)MM * FFD * 2);        // ffn hidden f16
  _Float16* Qh   = (_Float16*)carve((size_t)BB * HH * SS * HDD * 2);
  _Float16* Kh   = (_Float16*)carve((size_t)BB * HH * SS * HDD * 2);
  _Float16* Vt   = (_Float16*)carve((size_t)BB * HH * HDD * SS * 2);
  _Float16* Wh   = (_Float16*)carve((size_t)LL * LWH * 2);        // f16 W^T

  const dim3 blk256(256), blk128(128), blk32(32);

  // 1) convert + transpose all weights to f16 once per launch (deterministic)
  for (int l = 0; l < LL; ++l) {
    _Float16* base = Wh + (size_t)l * LWH;
    transpose_cvt<<<(DD * 3 * DD + 255) / 256, blk256, 0, stream>>>(
        in_w + (size_t)l * DD * 3 * DD, base, DD, 3 * DD);
    transpose_cvt<<<(DD * DD + 255) / 256, blk256, 0, stream>>>(
        out_w + (size_t)l * DD * DD, base + 196608, DD, DD);
    transpose_cvt<<<(DD * FFD + 255) / 256, blk256, 0, stream>>>(
        ffn_w1 + (size_t)l * DD * FFD, base + 262144, DD, FFD);
    transpose_cvt<<<(FFD * DD + 255) / 256, blk256, 0, stream>>>(
        ffn_w2 + (size_t)l * FFD * DD, base + 393216, FFD, DD);
  }

  // 2) embeddings + bias
  embed_kernel<<<MM, blk256, 0, stream>>>(cell_states, patch_emb, game_token, x);
  const int biasN = BB * HH * SS * SS;
  zero_f32<<<(biasN + 255) / 256, blk256, 0, stream>>>(bias, biasN);
  edge_scatter<<<(EE + 255) / 256, blk256, 0, stream>>>(edge_index, edge_type,
                                                        batch_vec, edge_emb, bias);

  // 3) transformer layers
  for (int l = 0; l < LL; ++l) {
    const _Float16* inT  = Wh + (size_t)l * LWH;
    const _Float16* outT = inT + 196608;
    const _Float16* w1T  = inT + 262144;
    const _Float16* w2T  = inT + 393216;

    ln_to_f16<<<MM, blk256, 0, stream>>>(x, ln1_g + l * DD, ln1_b + l * DD, a_h);
    gemm16<0><<<dim3(3 * DD / 64, MM / 64), blk128, 0, stream>>>(
        a_h, inT, in_b + (size_t)l * 3 * DD, DD, 0,
        nullptr, nullptr, Qh, Kh, Vt);
    attn_kernel<<<dim3(SS / 16, HH, BB), blk32, 0, stream>>>(Qh, Kh, Vt, bias, o_h);
    gemm16<1><<<dim3(DD / 64, MM / 64), blk128, 0, stream>>>(
        o_h, outT, out_b + (size_t)l * DD, DD, DD,
        x, nullptr, nullptr, nullptr, nullptr);
    ln_to_f16<<<MM, blk256, 0, stream>>>(x, ln2_g + l * DD, ln2_b + l * DD, a_h);
    gemm16<2><<<dim3(FFD / 64, MM / 64), blk128, 0, stream>>>(
        a_h, w1T, ffn_b1 + (size_t)l * FFD, DD, FFD,
        nullptr, fh, nullptr, nullptr, nullptr);
    gemm16<1><<<dim3(DD / 64, MM / 64), blk128, 0, stream>>>(
        fh, w2T, ffn_b2 + (size_t)l * DD, FFD, DD,
        x, nullptr, nullptr, nullptr, nullptr);
  }

  // 4) head
  head_kernel<<<1, blk256, 0, stream>>>(x, fc1_w, fc1_b, pol_w, pol_b,
                                        val_w, val_b, out);
}
